// MultiHeadSelfAttention_88407606821386
// MI455X (gfx1250) — matrix operand order repaired and checked by reading, not compiled, not run on hardware
//
#include <hip/hip_runtime.h>

typedef __bf16 bf16_t;
typedef __attribute__((ext_vector_type(16))) __bf16 v16bf;
typedef __attribute__((ext_vector_type(8)))  __bf16 v8bf;
typedef __attribute__((ext_vector_type(8)))  float  v8f;

static constexpr int Bc = 4, Sc = 2048, Dc = 1024, Hc = 8;
static constexpr int LDSS = 40;   // padded LDS row stride (32 K-elems + 8 pad) in bf16

// ---------------------------------------------------------------------------
// CDNA5 async global->LDS copy (ASYNCcnt-tracked), with portable fallback
// ---------------------------------------------------------------------------
#if defined(__has_builtin)
#  if __has_builtin(__builtin_amdgcn_global_load_async_to_lds_b128)
#    define CDNA5_HAS_ASYNC 1
#  endif
#endif
#ifndef CDNA5_HAS_ASYNC
#  define CDNA5_HAS_ASYNC 0
#endif

typedef int v4i_t __attribute__((vector_size(16)));
typedef __attribute__((address_space(1))) v4i_t as1_v4i;
typedef __attribute__((address_space(3))) v4i_t as3_v4i;

__device__ __forceinline__ void copy16B_g2lds(const bf16_t* g, bf16_t* l) {
#if CDNA5_HAS_ASYNC
  __builtin_amdgcn_global_load_async_to_lds_b128((as1_v4i*)g, (as3_v4i*)l, 0, 0);
#else
  *(v8bf*)l = *(const v8bf*)g;
#endif
}

template <int N>
__device__ __forceinline__ void wait_asynccnt_le() {
#if CDNA5_HAS_ASYNC
#  if __has_builtin(__builtin_amdgcn_s_wait_asynccnt)
  __builtin_amdgcn_s_wait_asynccnt(N);
#  else
  asm volatile("s_wait_asynccnt %0" ::"i"(N) : "memory");
#  endif
#endif
}

// ---------------------------------------------------------------------------
// fp32 -> bf16 conversion (grid-stride)
// ---------------------------------------------------------------------------
__global__ __launch_bounds__(256) void cvt_f32_to_bf16(const float* __restrict__ in,
                                                       bf16_t* __restrict__ out,
                                                       long long n) {
  long long i = (long long)blockIdx.x * blockDim.x + threadIdx.x;
  long long stride = (long long)gridDim.x * blockDim.x;
  for (; i < n; i += stride) out[i] = (bf16_t)in[i];
}

// ---------------------------------------------------------------------------
// fp32 [R][C] -> bf16 transposed [C][R] (tiled through LDS); blockIdx.z = slab
// ---------------------------------------------------------------------------
__global__ __launch_bounds__(256) void cvt_transpose_f32_to_bf16(
    const float* __restrict__ in, bf16_t* __restrict__ out, int R, int C) {
  __shared__ float tile[32][33];
  const long long slab = (long long)blockIdx.z * R * C;
  const int c0 = blockIdx.x * 32, r0 = blockIdx.y * 32;
  const int tx = threadIdx.x & 31, ty = threadIdx.x >> 5;  // 32 x 8
#pragma unroll
  for (int i = 0; i < 32; i += 8)
    tile[ty + i][tx] = in[slab + (long long)(r0 + ty + i) * C + c0 + tx];
  __syncthreads();
#pragma unroll
  for (int i = 0; i < 32; i += 8)
    out[slab + (long long)(c0 + ty + i) * R + r0 + tx] = (bf16_t)tile[tx][ty + i];
}

// ---------------------------------------------------------------------------
// bf16 WMMA GEMM, B always given transposed:  C[z] = scale * A[z] x Bt[z]^T (+resid)
// A: [M][K] row-major (lda);  Bt: [N][K] row-major (ldb).
// 256 threads = 8 wave32; WG tile 128x128; wave tile 64x32 = 4x2 wmma accs.
// Double-buffered LDS staging via async global->LDS b128 copies (ASYNCcnt).
// z offsets: off = (z/zdiv)*Hi + (z%zdiv)*Lo for A, Bt, C, resid.
// Epilogue: optional fp32 store, optional bf16 store (relu and/or transposed).
// ---------------------------------------------------------------------------
__global__ __launch_bounds__(256) void gemm_bt_wmma(
    const bf16_t* __restrict__ A, long long aHi, long long aLo, int lda,
    const bf16_t* __restrict__ Bt, long long bHi, long long bLo, int ldb,
    float* __restrict__ outF32, bf16_t* __restrict__ outBf16,
    long long cHi, long long cLo, int ldc, int ldcT, int transBf16, int reluBf16,
    const float* __restrict__ resid, long long rHi, long long rLo, int ldr,
    int zdiv, float scale, int K) {
  __shared__ bf16_t As[2][128 * LDSS];
  __shared__ bf16_t Bs[2][128 * LDSS];

  const int tid = threadIdx.x;
  const int z = blockIdx.z;
  const int zh = z / zdiv, zl = z % zdiv;
  const bf16_t* Ag = A + (long long)zh * aHi + (long long)zl * aLo;
  const bf16_t* Bg = Bt + (long long)zh * bHi + (long long)zl * bLo;
  const long long cOff = (long long)zh * cHi + (long long)zl * cLo;
  const long long rOff = (long long)zh * rHi + (long long)zl * rLo;

  const int bm = blockIdx.x * 128;
  const int bn = blockIdx.y * 128;

  const int wave = tid >> 5;
  const int lane = tid & 31;
  const int wm = (wave & 1) * 64;
  const int wn = (wave >> 1) * 32;

  v8f acc[4][2] = {};

  const int sm  = tid >> 1;        // staging row 0..127
  const int skh = (tid & 1) * 16;  // staging K half (elems)

  auto stage = [&](bf16_t* Ad, bf16_t* Bd, int k0) {
    const bf16_t* ga = Ag + (long long)(bm + sm) * lda + k0 + skh;
    const bf16_t* gb = Bg + (long long)(bn + sm) * ldb + k0 + skh;
    bf16_t* la = Ad + sm * LDSS + skh;
    bf16_t* lb = Bd + sm * LDSS + skh;
    copy16B_g2lds(ga, la);
    copy16B_g2lds(ga + 8, la + 8);
    copy16B_g2lds(gb, lb);
    copy16B_g2lds(gb + 8, lb + 8);
  };

  stage(As[0], Bs[0], 0);

  for (int k0 = 0; k0 < K; k0 += 32) {
    const int cur = (k0 >> 5) & 1;
    const bool more = (k0 + 32) < K;
    if (more) {
      stage(As[cur ^ 1], Bs[cur ^ 1], k0 + 32);
      wait_asynccnt_le<4>();  // our 4 copies into buf[cur] are done (in-order)
    } else {
      wait_asynccnt_le<0>();
    }
    __syncthreads();

    // A fragment (16-bit A layout: lanes 0-15 K{0-7,16-23}; lanes 16-31 K{8-15,24-31})
    v16bf af[4];
#pragma unroll
    for (int mi = 0; mi < 4; mi++) {
      const bf16_t* p = As[cur] + (wm + mi * 16 + (lane & 15)) * LDSS + (lane >> 4) * 8;
      v8bf lo = *(const v8bf*)p;
      v8bf hi = *(const v8bf*)(p + 16);
      af[mi] = __builtin_shufflevector(lo, hi, 0, 1, 2, 3, 4, 5, 6, 7, 8, 9, 10, 11, 12, 13, 14, 15);
    }
    // B fragment (lanes 0-15: col n, K{0-7,16-23}; lanes 16-31: col n, K{8-15,24-31})
    v16bf bfr[2];
#pragma unroll
    for (int ni = 0; ni < 2; ni++) {
      const bf16_t* p = Bs[cur] + (wn + ni * 16 + (lane & 15)) * LDSS + (lane >> 4) * 8;
      v8bf lo = *(const v8bf*)p;
      v8bf hi = *(const v8bf*)(p + 16);
      bfr[ni] = __builtin_shufflevector(lo, hi, 0, 1, 2, 3, 4, 5, 6, 7, 8, 9, 10, 11, 12, 13, 14, 15);
    }
#pragma unroll
    for (int mi = 0; mi < 4; mi++)
#pragma unroll
      for (int ni = 0; ni < 2; ni++)
        acc[mi][ni] = __builtin_amdgcn_wmma_f32_16x16x32_bf16(
            false, af[mi], false, bfr[ni], (short)0, acc[mi][ni], false, false);
    __syncthreads();
  }

  // epilogue: C layout VGPR r -> M = r + 8*(lane>=16), N = lane&15
  const int colLane = lane & 15;
  const int hi8 = (lane >> 4) * 8;
#pragma unroll
  for (int mi = 0; mi < 4; mi++) {
#pragma unroll
    for (int ni = 0; ni < 2; ni++) {
      const int col = bn + wn + ni * 16 + colLane;
#pragma unroll
      for (int r = 0; r < 8; r++) {
        const int row = bm + wm + mi * 16 + r + hi8;
        float v = acc[mi][ni][r] * scale;
        if (resid) v += resid[rOff + (long long)row * ldr + col];
        if (outF32) outF32[cOff + (long long)row * ldc + col] = v;
        if (outBf16) {
          const float o = reluBf16 ? fmaxf(v, 0.f) : v;
          if (transBf16)
            outBf16[cOff + (long long)col * ldcT + row] = (bf16_t)o;
          else
            outBf16[cOff + (long long)row * ldc + col] = (bf16_t)o;
        }
      }
    }
  }
}

// ---------------------------------------------------------------------------
// Row softmax over length-2048 rows: fp32 in -> bf16 probabilities out
// ---------------------------------------------------------------------------
__global__ __launch_bounds__(256) void softmax_rows(const float* __restrict__ scores,
                                                    bf16_t* __restrict__ probs, int S) {
  __shared__ float red[256];
  const long long base = (long long)blockIdx.x * S;
  const int tid = threadIdx.x;
  float v[8];
  float mx = -3.4e38f;
#pragma unroll
  for (int i = 0; i < 8; i++) {
    v[i] = scores[base + tid + i * 256];
    mx = fmaxf(mx, v[i]);
  }
  red[tid] = mx;
  __syncthreads();
  for (int s = 128; s > 0; s >>= 1) {
    if (tid < s) red[tid] = fmaxf(red[tid], red[tid + s]);
    __syncthreads();
  }
  mx = red[0];
  __syncthreads();
  float sum = 0.f;
#pragma unroll
  for (int i = 0; i < 8; i++) {
    v[i] = __expf(v[i] - mx);
    sum += v[i];
  }
  red[tid] = sum;
  __syncthreads();
  for (int s = 128; s > 0; s >>= 1) {
    if (tid < s) red[tid] += red[tid + s];
    __syncthreads();
  }
  const float inv = 1.f / red[0];
#pragma unroll
  for (int i = 0; i < 8; i++) probs[base + tid + i * 256] = (bf16_t)(v[i] * inv);
}

// ---------------------------------------------------------------------------
// Row layernorm: one block per row; writes fp32 and/or bf16
// ---------------------------------------------------------------------------
__global__ __launch_bounds__(256) void layernorm_rows(const float* __restrict__ x,
                                                      const float* __restrict__ g,
                                                      const float* __restrict__ b,
                                                      float* __restrict__ outF32,
                                                      bf16_t* __restrict__ outBf16,
                                                      int Dlen, float eps) {
  __shared__ float red[256];
  const long long base = (long long)blockIdx.x * Dlen;
  const int tid = threadIdx.x;
  float sum = 0.f, sq = 0.f;
  for (int i = tid; i < Dlen; i += 256) {
    float t = x[base + i];
    sum += t;
    sq += t * t;
  }
  red[tid] = sum;
  __syncthreads();
  for (int s = 128; s > 0; s >>= 1) {
    if (tid < s) red[tid] += red[tid + s];
    __syncthreads();
  }
  sum = red[0];
  __syncthreads();
  red[tid] = sq;
  __syncthreads();
  for (int s = 128; s > 0; s >>= 1) {
    if (tid < s) red[tid] += red[tid + s];
    __syncthreads();
  }
  sq = red[0];
  const float m = sum / (float)Dlen;
  const float var = sq / (float)Dlen - m * m;
  const float inv = rsqrtf(var + eps);
  for (int i = tid; i < Dlen; i += 256) {
    float t = (x[base + i] - m) * inv * g[i] + b[i];
    if (outF32) outF32[base + i] = t;
    if (outBf16) outBf16[base + i] = (bf16_t)t;
  }
}

// ---------------------------------------------------------------------------
// Host-side orchestration
// ---------------------------------------------------------------------------
extern "C" void kernel_launch(void* const* d_in, const int* in_sizes, int n_in,
                              void* d_out, int out_size, void* d_ws, size_t ws_size,
                              hipStream_t stream) {
  (void)in_sizes; (void)n_in; (void)out_size; (void)ws_size;
  const float* x  = (const float*)d_in[0];
  const float* Wk = (const float*)d_in[1];
  const float* Wq = (const float*)d_in[2];
  const float* Wv = (const float*)d_in[3];
  const float* Wh = (const float*)d_in[4];
  const float* Wl = (const float*)d_in[5];
  const float* g1 = (const float*)d_in[6];
  const float* b1 = (const float*)d_in[7];
  const float* g2 = (const float*)d_in[8];
  const float* b2 = (const float*)d_in[9];
  float* out = (float*)d_out;

  const long long nX  = (long long)Bc * Sc * Dc;  // 8.39M (= D*B*S)
  const long long nW  = (long long)Dc * Dc;
  const long long nWH = (long long)Hc * Dc * Dc;
  const long long nSS = (long long)Sc * Sc;
  const int Z = Hc * Bc;  // 32, z = h*B + b

  char* ws = (char*)d_ws;
  size_t off = 0;
  auto take = [&](size_t bytes) -> char* {
    char* p = ws + off;
    off = (off + bytes + 255) & ~(size_t)255;
    return p;
  };
  bf16_t* xbf    = (bf16_t*)take((size_t)nX * 2);
  bf16_t* wkT    = (bf16_t*)take((size_t)nWH * 2);  // [h][e][d]
  bf16_t* wqT    = (bf16_t*)take((size_t)nWH * 2);
  bf16_t* wvT    = (bf16_t*)take((size_t)nWH * 2);
  bf16_t* whT    = (bf16_t*)take((size_t)nWH * 2);  // [D][H*D]
  bf16_t* linT   = (bf16_t*)take((size_t)nW * 2);
  bf16_t* kq     = (bf16_t*)take((size_t)2 * Hc * nX * 2);  // [op(k,q)][h][b][s][d]
  bf16_t* vvT    = (bf16_t*)take((size_t)Hc * nX * 2);      // [h][d][b*S+s]
  float*  scores = (float*)take((size_t)Z * nSS * 4);       // [z][s][t]
  bf16_t* probs  = (bf16_t*)take((size_t)Z * nSS * 2);
  float*  xcat   = (float*)take((size_t)Bc * Sc * Hc * Dc * 4);  // [b][s][h*D+d]
  bf16_t* xn     = (bf16_t*)take((size_t)Bc * Sc * Hc * Dc * 2);
  float*  xt     = (float*)take((size_t)nX * 4);
  bf16_t* rl     = (bf16_t*)take((size_t)nX * 2);
  float*  y      = (float*)take((size_t)nX * 4);

  dim3 blk(256);

  // ---- 1. conversions (weights transposed once so every GEMM is BT mode) ----
  cvt_f32_to_bf16<<<2048, blk, 0, stream>>>(x, xbf, nX);
  {
    dim3 gT(Dc / 32, Dc / 32, Hc);
    cvt_transpose_f32_to_bf16<<<gT, blk, 0, stream>>>(Wk, wkT, Dc, Dc);
    cvt_transpose_f32_to_bf16<<<gT, blk, 0, stream>>>(Wq, wqT, Dc, Dc);
    cvt_transpose_f32_to_bf16<<<gT, blk, 0, stream>>>(Wv, wvT, Dc, Dc);
    dim3 gH(Dc / 32, (Hc * Dc) / 32, 1);
    cvt_transpose_f32_to_bf16<<<gH, blk, 0, stream>>>(Wh, whT, Hc * Dc, Dc);
    dim3 gL(Dc / 32, Dc / 32, 1);
    cvt_transpose_f32_to_bf16<<<gL, blk, 0, stream>>>(Wl, linT, Dc, Dc);
  }

  // ---- 2. QKV projections (z = head). K,Q stored [h][b][s][d]; V stored transposed ----
  {
    dim3 grd((Bc * Sc) / 128, Dc / 128, Hc);
    // K
    gemm_bt_wmma<<<grd, blk, 0, stream>>>(
        xbf, 0LL, 0LL, Dc, wkT, nW, 0LL, Dc,
        (float*)nullptr, kq, nX, 0LL, Dc, Dc, 0, 0,
        (const float*)nullptr, 0LL, 0LL, Dc, 1, 1.0f, Dc);
    // Q
    gemm_bt_wmma<<<grd, blk, 0, stream>>>(
        xbf, 0LL, 0LL, Dc, wqT, nW, 0LL, Dc,
        (float*)nullptr, kq + Hc * nX, nX, 0LL, Dc, Dc, 0, 0,
        (const float*)nullptr, 0LL, 0LL, Dc, 1, 1.0f, Dc);
    // V (transposed store: vvT[h][d][b*S+s])
    gemm_bt_wmma<<<grd, blk, 0, stream>>>(
        xbf, 0LL, 0LL, Dc, wvT, nW, 0LL, Dc,
        (float*)nullptr, vvT, nX, 0LL, Dc, Bc * Sc, 1, 0,
        (const float*)nullptr, 0LL, 0LL, Dc, 1, 1.0f, Dc);
  }

  // ---- 3. scores[z] = (K[z] x Q[z]^T) / sqrt(D); z = h*B + b ----
  {
    dim3 grd(Sc / 128, Sc / 128, Z);
    gemm_bt_wmma<<<grd, blk, 0, stream>>>(
        kq, (long long)Sc * Dc, 0LL, Dc,
        kq + Hc * nX, (long long)Sc * Dc, 0LL, Dc,
        scores, (bf16_t*)nullptr, nSS, 0LL, Sc, Sc, 0, 0,
        (const float*)nullptr, 0LL, 0LL, Dc, 1, 0.03125f /* 1/sqrt(1024) */, Dc);
  }

  // ---- 4. row softmax -> bf16 probs ----
  softmax_rows<<<Z * Sc, blk, 0, stream>>>(scores, probs, Sc);

  // ---- 5. attn out: probs x vvT^T + x (per-head residual) scattered into xcat ----
  {
    dim3 grd(Sc / 128, Dc / 128, Z);  // z = h*B + b, zdiv = B
    gemm_bt_wmma<<<grd, blk, 0, stream>>>(
        probs, (long long)Bc * nSS, nSS, Sc,
        vvT, (long long)Dc * Bc * Sc, (long long)Sc, Bc * Sc,
        xcat, (bf16_t*)nullptr,
        (long long)Dc /* per h: col offset */, (long long)Sc * Hc * Dc /* per b */,
        Hc * Dc, Hc * Dc, 0, 0,
        x, 0LL, (long long)Sc * Dc, Dc,
        Bc, 1.0f, Sc);
  }

  // ---- 6. layernorm over H*D = 8192 features -> bf16 xn ----
  layernorm_rows<<<Bc * Sc, blk, 0, stream>>>(xcat, g1, b1, (float*)nullptr, xn, Hc * Dc, 1e-3f);

  // ---- 7. xt = xn x w_heads (fp32 xt + relu bf16) ----
  {
    dim3 grd((Bc * Sc) / 128, Dc / 128, 1);
    gemm_bt_wmma<<<grd, blk, 0, stream>>>(
        xn, 0LL, 0LL, Hc * Dc, whT, 0LL, 0LL, Hc * Dc,
        xt, rl, 0LL, 0LL, Dc, Dc, 0, 1 /* relu into bf16 */,
        (const float*)nullptr, 0LL, 0LL, Dc, 1, 1.0f, Hc * Dc);
  }

  // ---- 8. y = relu(xt) x lin + xt ----
  {
    dim3 grd((Bc * Sc) / 128, Dc / 128, 1);
    gemm_bt_wmma<<<grd, blk, 0, stream>>>(
        rl, 0LL, 0LL, Dc, linT, 0LL, 0LL, Dc,
        y, (bf16_t*)nullptr, 0LL, 0LL, Dc, Dc, 0, 0,
        xt, 0LL, 0LL, Dc, 1, 1.0f, Dc);
  }

  // ---- 9. final layernorm over D -> d_out (fp32) ----
  layernorm_rows<<<Bc * Sc, blk, 0, stream>>>(y, g2, b2, out, (bf16_t*)nullptr, Dc, 1e-3f);
}
